// TipAdapterPlus_31069793419865
// MI455X (gfx1250) — compile-verified, matrix operand order checked by reading, not run on hardware
//
#include <hip/hip_runtime.h>

// Problem sizes (fixed by the reference)
#define B_   4096
#define D_   1024
#define NS_  16000
#define NSP_ 16128    // NS padded to multiple of 256 (zero-padded in workspace)
#define C_   1000
#define CT_  1024     // C padded to WMMA-friendly multiple of 64
#define H_   256

typedef __attribute__((ext_vector_type(16))) __bf16 v16bf;
typedef __attribute__((ext_vector_type(8)))  float  v8f;

union FragBF {
  v16bf v;
  unsigned short h[16];
  unsigned int   u[8];
  uint4          q[2];
};

// round-to-nearest-even float -> bf16 (bit pattern)
static __device__ __forceinline__ unsigned short f2bf(float f) {
  union { float f; unsigned u; } x; x.f = f;
  unsigned r = x.u + 0x7FFFu + ((x.u >> 16) & 1u);
  return (unsigned short)(r >> 16);
}

// ---------------------------------------------------------------------------
// A-fragment (16xK tile at k0) from bf16 row-major LDS, ld = halfs per row.
// ISA 7.12.2 16-bit A layout: lanes 0-15 -> M=lane, K {k0..+7, k0+16..+23};
// lanes 16-31 -> M=lane-16, K {k0+8..+15, k0+24..+31}.
static __device__ __forceinline__ v16bf load_a_lds(const __bf16* base, int ld,
                                                   int k0, int lane) {
  FragBF a;
  int m  = lane & 15;
  int kb = k0 + ((lane >> 4) << 3);
  const __bf16* p = base + m * ld + kb;
  a.q[0] = *reinterpret_cast<const uint4*>(p);        // ds_load_b128
  a.q[1] = *reinterpret_cast<const uint4*>(p + 16);
  return a.v;
}

// B-fragment (K=32 x N=16): B[k][n] = base[(n0+n)*ld + k0+k]
// (column per lane, K contiguous in memory -> 2x global_load_b128 of bf16)
static __device__ __forceinline__ v16bf load_b_bf16(const __bf16* __restrict__ base,
                                                    size_t ld, int n0, int k0,
                                                    int lane) {
  FragBF b;
  int n  = lane & 15;
  int kb = k0 + ((lane >> 4) << 3);
  const __bf16* p = base + (size_t)(n0 + n) * ld + kb;
  b.q[0] = *reinterpret_cast<const uint4*>(p);
  b.q[1] = *reinterpret_cast<const uint4*>(p + 16);
  return b.v;
}

// ---------------------------------------------------------------------------
// Prep kernel A: keys f32 [NS][D] -> bf16 [NSP][D] (rows >= NS zeroed)
__launch_bounds__(256)
__global__ void cvt_keys_kernel(const float* __restrict__ keys,
                                __bf16* __restrict__ kb) {
  size_t e = ((size_t)blockIdx.x * 256 + threadIdx.x) * 4;
  if (e >= (size_t)NSP_ * D_) return;
  uint2 pk = {0u, 0u};
  if ((e >> 10) < (size_t)NS_) {   // row = e / D_
    float4 f = *reinterpret_cast<const float4*>(keys + e);
    pk.x = (unsigned)f2bf(f.x) | ((unsigned)f2bf(f.y) << 16);
    pk.y = (unsigned)f2bf(f.z) | ((unsigned)f2bf(f.w) << 16);
  }
  *reinterpret_cast<uint2*>(kb + e) = pk;
}

// Prep kernel B: values f32 [NS][C] -> bf16 transposed [CT][NSP] (zero-padded)
__launch_bounds__(256)
__global__ void transpose_vals_kernel(const float* __restrict__ vals,
                                      __bf16* __restrict__ vt) {
  __shared__ float tile[32][33];
  const int c0  = blockIdx.x * 32;   // 0..992
  const int ns0 = blockIdx.y * 32;   // 0..16096
  const int tx  = threadIdx.x & 31;
  const int ty  = threadIdx.x >> 5;  // 0..7
  for (int r = ty; r < 32; r += 8) {
    int c  = c0 + tx;
    int ns = ns0 + r;
    tile[r][tx] = (c < C_ && ns < NS_) ? vals[(size_t)ns * C_ + c] : 0.0f;
  }
  __syncthreads();
  for (int r = ty; r < 32; r += 8) {
    reinterpret_cast<unsigned short&>(vt[(size_t)(c0 + r) * NSP_ + ns0 + tx]) =
        f2bf(tile[tx][r]);
  }
}

// ---------------------------------------------------------------------------
// Kernel 1: gate MLP -> alpha, beta per batch row (tiny: ~2 GFLOP total)
__launch_bounds__(256, 4)
__global__ void gate_mlp_kernel(const float* __restrict__ q,
                                const float* __restrict__ fc1_w,
                                const float* __restrict__ fc1_b,
                                const float* __restrict__ fc2_w,
                                const float* __restrict__ fc2_b,
                                float* __restrict__ alpha_ws,
                                float* __restrict__ beta_ws,
                                float* __restrict__ out_alpha,
                                float* __restrict__ out_beta) {
  __shared__ float qrow[D_];
  __shared__ float red0[256];
  __shared__ float red1[256];
  const int b   = blockIdx.x;
  const int tid = threadIdx.x;

  for (int d = tid; d < D_; d += 256) qrow[d] = q[(size_t)b * D_ + d];
  __syncthreads();

  const float* w = fc1_w + (size_t)tid * D_;   // tid == hidden unit (H_==256)
  float acc = 0.f;
#pragma unroll 4
  for (int d = 0; d < D_; d += 4) {
    float4 wv = *reinterpret_cast<const float4*>(w + d);
    acc += wv.x * qrow[d] + wv.y * qrow[d + 1] + wv.z * qrow[d + 2] + wv.w * qrow[d + 3];
  }
  float x = acc + fc1_b[tid];
  x = x > 0.f ? x : 0.f;                       // relu
  red0[tid] = x * fc2_w[tid];                  // fc2_w[0][h]
  red1[tid] = x * fc2_w[H_ + tid];             // fc2_w[1][h]
  __syncthreads();
  for (int s = 128; s > 0; s >>= 1) {
    if (tid < s) { red0[tid] += red0[tid + s]; red1[tid] += red1[tid + s]; }
    __syncthreads();
  }
  if (tid == 0) {
    float g0 = red0[0] + fc2_b[0];
    float g1 = red1[0] + fc2_b[1];
    float alpha = 1.f / (1.f + __expf(-g0));                       // sigmoid
    float beta  = (g1 > 20.f ? g1 : log1pf(__expf(g1))) + 0.001f;  // softplus + eps
    alpha_ws[b] = alpha;  beta_ws[b] = beta;
    out_alpha[b] = alpha; out_beta[b] = beta;
  }
}

// ---------------------------------------------------------------------------
// Kernel 2: fused  sim = q@keys^T ; aff = exp(beta*sim) ; c = aff@values ;
//           logits = ((1-alpha)*z + alpha*c) * post_scale
// Block: 32 batch rows (2 M-subtiles), 16 waves (512 thr). NS chunk = 256:
//   GEMM1: wave w -> sim [2 x 16x16] for ns cols [ns0+w*16,+16)  (64 WMMA)
//   GEMM2: wave w -> 64-wide C slice, K=256, both subtiles        (64 WMMA)
// Every B fragment (keys & values) feeds 2 WMMAs -> halved L2 traffic/row.
__launch_bounds__(512, 1)
__global__ void fused_cache_kernel(const float* __restrict__ q,
                                   const float* __restrict__ z,
                                   const __bf16* __restrict__ keys_bf, // [NSP][D]
                                   const __bf16* __restrict__ vals_t,  // [CT][NSP]
                                   const float* __restrict__ alpha_ws,
                                   const float* __restrict__ beta_ws,
                                   const float* __restrict__ post_scale,
                                   float* __restrict__ out) {
  // padded strides keep ds_load_b128 16B-aligned and spread across 64 banks
  __shared__ __bf16 qs[32 * 1032];     // q tile bf16 (32 rows), stride 1032
  __shared__ __bf16 affs[32 * 264];    // affinity chunk [32 x 256], stride 264
  __shared__ float alpha_s[32];
  __shared__ float beta_s[32];

  const int tid  = threadIdx.x;
  const int lane = tid & 31;
  const int wave = tid >> 5;                 // 0..15
  const int b0   = blockIdx.x * 32;

  // stage q[b0:b0+32, :] as bf16 into LDS (pairs -> packed dword stores)
  for (int i = tid; i < (32 * 1024) / 2; i += 512) {
    int e = i * 2;
    int r = e >> 10;
    int c = e & 1023;
    float2 f = *reinterpret_cast<const float2*>(q + (size_t)(b0 + r) * D_ + c);
    unsigned pk = (unsigned)f2bf(f.x) | ((unsigned)f2bf(f.y) << 16);
    *reinterpret_cast<unsigned*>(&qs[r * 1032 + c]) = pk;
  }
  if (tid < 32) { alpha_s[tid] = alpha_ws[b0 + tid]; beta_s[tid] = beta_ws[b0 + tid]; }
  __syncthreads();

  v8f acc[2][4];
#pragma unroll
  for (int mi = 0; mi < 2; ++mi)
#pragma unroll
    for (int t = 0; t < 4; ++t) acc[mi][t] = v8f{};

  const int nloc = lane & 15;
  const int mh   = (lane >> 4) << 3;

  for (int ns0 = 0; ns0 < NSP_; ns0 += 256) {
    // ---- GEMM1: sim tiles (both M-subtiles) for cols [ns0+wave*16, +16) ----
    const int nsw = ns0 + wave * 16;
    v8f sim0 = v8f{};
    v8f sim1 = v8f{};
#pragma unroll 4
    for (int d0 = 0; d0 < D_; d0 += 32) {
      v16bf bk = load_b_bf16(keys_bf, (size_t)D_, nsw, d0, lane);  // shared B
      v16bf a0 = load_a_lds(qs, 1032, d0, lane);                   // rows 0..15
      v16bf a1 = load_a_lds(qs + 16 * 1032, 1032, d0, lane);       // rows 16..31
      sim0 = __builtin_amdgcn_wmma_f32_16x16x32_bf16(
          false, a0, false, bk, (short)0, sim0, false, false);
      sim1 = __builtin_amdgcn_wmma_f32_16x16x32_bf16(
          false, a1, false, bk, (short)0, sim1, false, false);
    }
    // prefetch next chunk's keys rows (global_prefetch_b8)
    if (ns0 + 256 < NSP_)
      __builtin_prefetch(keys_bf + (size_t)(nsw + 256 + nloc) * D_, 0, 3);

    // ---- epilogue of GEMM1: affinity = exp(beta * sim) -> bf16 LDS ----
#pragma unroll
    for (int r = 0; r < 8; ++r) {
      int m = mh + r;                        // C/D layout: M = r + 8*(lane>=16)
      float a0 = __expf(beta_s[m] * sim0[r]);
      float a1 = __expf(beta_s[16 + m] * sim1[r]);
      reinterpret_cast<unsigned short&>(
          affs[m * 264 + wave * 16 + nloc]) = f2bf(a0);
      reinterpret_cast<unsigned short&>(
          affs[(16 + m) * 264 + wave * 16 + nloc]) = f2bf(a1);
    }
    __syncthreads();

    // ---- GEMM2: acc[32 x 64 C-slice] += aff[32x256] @ values[256 x C] ----
    const int cbase = wave * 64;
#pragma unroll 2
    for (int kk = 0; kk < 8; ++kk) {
      v16bf a0 = load_a_lds(affs, 264, kk * 32, lane);
      v16bf a1 = load_a_lds(affs + 16 * 264, 264, kk * 32, lane);
#pragma unroll
      for (int t = 0; t < 4; ++t) {
        // B[k][n] = vals_t[(cbase+t*16+n)*NSP + ns0+kk*32+k]   (shared B)
        v16bf bv = load_b_bf16(vals_t, (size_t)NSP_, cbase + t * 16,
                               ns0 + kk * 32, lane);
        acc[0][t] = __builtin_amdgcn_wmma_f32_16x16x32_bf16(
            false, a0, false, bv, (short)0, acc[0][t], false, false);
        acc[1][t] = __builtin_amdgcn_wmma_f32_16x16x32_bf16(
            false, a1, false, bv, (short)0, acc[1][t], false, false);
      }
    }
    __syncthreads();   // protect affs WAR before next chunk's writes
  }

  // ---- fusion epilogue: logits = ((1-alpha)*z + alpha*c) * post_scale ----
  const float ps = post_scale[0];
#pragma unroll
  for (int mi = 0; mi < 2; ++mi) {
#pragma unroll
    for (int t = 0; t < 4; ++t) {
      int c = wave * 64 + t * 16 + nloc;
      if (c >= C_) continue;
#pragma unroll
      for (int r = 0; r < 8; ++r) {
        int m = mi * 16 + mh + r;
        int b = b0 + m;
        float al = alpha_s[m];
        float zv = z[(size_t)b * C_ + c];
        out[(size_t)b * C_ + c] = ((1.f - al) * zv + al * acc[mi][t][r]) * ps;
      }
    }
  }
}

// ---------------------------------------------------------------------------
extern "C" void kernel_launch(void* const* d_in, const int* in_sizes, int n_in,
                              void* d_out, int out_size, void* d_ws, size_t ws_size,
                              hipStream_t stream) {
  const float* q      = (const float*)d_in[0];
  const float* z      = (const float*)d_in[1];
  const float* keys   = (const float*)d_in[2];
  const float* values = (const float*)d_in[3];
  const float* fc1_w  = (const float*)d_in[4];
  const float* fc1_b  = (const float*)d_in[5];
  const float* fc2_w  = (const float*)d_in[6];
  const float* fc2_b  = (const float*)d_in[7];
  const float* ps     = (const float*)d_in[8];

  float* out = (float*)d_out;

  // workspace layout:
  //   alpha [B] f32 | beta [B] f32 | keys_bf [NSP*D] bf16 | vals_t [CT*NSP] bf16
  char* wsp = (char*)d_ws;
  float*  alpha_ws = (float*)wsp;
  float*  beta_ws  = alpha_ws + B_;
  __bf16* keys_bf  = (__bf16*)(wsp + 2 * B_ * sizeof(float));
  __bf16* vals_t   = keys_bf + (size_t)NSP_ * D_;

  // output tuple layout: logits [B*C], alpha [B], beta [B]
  float* out_alpha = out + (size_t)B_ * C_;
  float* out_beta  = out_alpha + B_;

  // prep: bf16 copies (keys row-major zero-padded, values transposed+padded)
  cvt_keys_kernel<<<((size_t)NSP_ * D_ / 4 + 255) / 256, 256, 0, stream>>>(keys, keys_bf);
  transpose_vals_kernel<<<dim3(CT_ / 32, NSP_ / 32), 256, 0, stream>>>(values, vals_t);

  gate_mlp_kernel<<<B_, 256, 0, stream>>>(q, fc1_w, fc1_b, fc2_w, fc2_b,
                                          alpha_ws, beta_ws, out_alpha, out_beta);
  fused_cache_kernel<<<B_ / 32, 512, 0, stream>>>(q, z, keys_bf, vals_t,
                                                  alpha_ws, beta_ws, ps, out);
}
